// ComplexCNN2d_70832600645950
// MI455X (gfx1250) — matrix operand order repaired and checked by reading, not compiled, not run on hardware
//
#include <hip/hip_runtime.h>

// ---------------- problem constants ----------------
#define B_      32
#define CIN_    2
#define COUT_   64
#define H_      128
#define W_      128
#define KS_     5
#define PAD_    2
#define HW_     (H_*W_)
#define NPIX_   (B_*HW_)            // 524288 output pixels
#define KRED_   128                 // reduction 2*CIN*K*K = 100, padded to 128
#define ASTR_   136                 // padded LDS row stride (bf16 elems) for bank spread
#define WAVES_  8                   // waves per block (wave32)
#define NTILE_  (NPIX_/16)          // 32768 16-pixel tiles
#define BLOCKS1_ (NTILE_/WAVES_)    // 4096 blocks
#define IMAG_OFF_ (B_*COUT_*HW_)    // 33554432 floats
#define COEF_OFF_ (BLOCKS1_*256)    // float offset of BN coef region in ws
#define TSTR_   17                  // transpose buffer row stride (float2), padded

typedef __attribute__((ext_vector_type(16))) __bf16 bf16x16;
typedef __attribute__((ext_vector_type(8)))  __bf16 bf16x8;
typedef __attribute__((ext_vector_type(2)))  __bf16 bf16x2;
typedef __attribute__((ext_vector_type(8)))  float  v8f;

// ============================================================================
// Pass 1: complex conv as implicit GEMM with v_wmma_f32_16x16x32_bf16.
// Writes un-normalized Y (real+imag, with bias) to d_out and deterministic
// per-block BN partial sums (sum, sum^2 per channel) to d_ws.
// ============================================================================
extern "C" __global__ __launch_bounds__(256)
void cplx_conv_wmma(const float* __restrict__ Xr, const float* __restrict__ Xi,
                    const float* __restrict__ Wr, const float* __restrict__ Wi,
                    const float* __restrict__ br, const float* __restrict__ bi,
                    float* __restrict__ out, float* __restrict__ partials)
{
    extern __shared__ char smem[];
    __bf16* Ar = (__bf16*)smem;                 // [64][ASTR_]  A_r = [Wr | -Wi | 0]
    __bf16* Ai = Ar + COUT_*ASTR_;              // [64][ASTR_]  A_i = [Wi |  Wr | 0]
    __bf16* Bc = Ai + COUT_*ASTR_;              // [8 waves][16 cols][ASTR_] im2col
    float*  WS = (float*)(Bc + WAVES_*16*ASTR_);// [8][64][4] per-wave ch stats
    int*    LUT = (int*)(WS + WAVES_*COUT_*4);  // [128] kk decode
    float2* TR  = (float2*)(LUT + KRED_);       // [8][16ch][TSTR_] (yr,yi) transpose

    const int tid  = threadIdx.x;
    const int lane = tid & 31;
    const int wave = tid >> 5;

    // ---- build bf16 A matrices, two K-elements per store (b64 in, b32 out) ----
    for (int idx = tid; idx < COUT_*(KRED_/2); idx += 256) {
        int ch = idx >> 6;
        int kk = (idx & 63) << 1;                // even; r/i boundary (50) is even
        float ar0 = 0.f, ar1 = 0.f, ai0 = 0.f, ai1 = 0.f;
        if (kk < 50) {
            ar0 =  Wr[ch*50 + kk];     ar1 =  Wr[ch*50 + kk + 1];
            ai0 =  Wi[ch*50 + kk];     ai1 =  Wi[ch*50 + kk + 1];
        } else if (kk < 100) {
            int j = kk - 50;
            ar0 = -Wi[ch*50 + j];      ar1 = -Wi[ch*50 + j + 1];
            ai0 =  Wr[ch*50 + j];      ai1 =  Wr[ch*50 + j + 1];
        }
        bf16x2 pr; pr[0] = (__bf16)ar0; pr[1] = (__bf16)ar1;
        bf16x2 pi; pi[0] = (__bf16)ai0; pi[1] = (__bf16)ai1;
        *(bf16x2*)(Ar + ch*ASTR_ + kk) = pr;
        *(bf16x2*)(Ai + ch*ASTR_ + kk) = pi;
    }
    // ---- build kk-decode LUT: delta[15:0] | rr[18:16] | ss[22:20] | sel[24] ----
    for (int kk = tid; kk < KRED_; kk += 256) {
        int e = (int)0x80000000;    // invalid (zero-fill)
        if (kk < 100) {
            int sel = (kk >= 50) ? 1 : 0;
            int k2 = sel ? kk - 50 : kk;
            int c = k2 / 25, rm = k2 % 25;
            int rr = rm / 5, ss = rm % 5;
            e = (c*HW_ + rr*W_ + ss) | (rr << 16) | (ss << 20) | (sel << 24);
        }
        LUT[kk] = e;
    }
    __syncthreads();

    // ---- build this wave's im2col B tile (column-major: K contiguous) ----
    const int tile = blockIdx.x * WAVES_ + wave;     // 16 consecutive pixels in W
    const int w0 = (tile & 7) << 4;
    const int h  = (tile >> 3) & (H_ - 1);
    const int b  = tile >> 10;
    __bf16* Bw = Bc + wave * 16 * ASTR_;
    const int basePix = (b*CIN_)*HW_ + (h - PAD_)*W_ + (w0 - PAD_);
    for (int idx = lane; idx < 16 * (KRED_/2); idx += 32) {
        int n  = idx & 15;
        int kk = (idx >> 4) << 1;                // even K pair
        int2 e2 = *(const int2*)(LUT + kk);
        float v0 = 0.f, v1 = 0.f;
        if (e2.x >= 0) {
            int rr = (e2.x >> 16) & 7, ss = (e2.x >> 20) & 7;
            int ih = h - PAD_ + rr, iw = w0 + n - PAD_ + ss;
            if ((unsigned)ih < (unsigned)H_ && (unsigned)iw < (unsigned)W_) {
                const float* src = (e2.x & (1 << 24)) ? Xi : Xr;
                v0 = src[basePix + (e2.x & 0xffff) + n];
            }
        }
        if (e2.y >= 0) {
            int rr = (e2.y >> 16) & 7, ss = (e2.y >> 20) & 7;
            int ih = h - PAD_ + rr, iw = w0 + n - PAD_ + ss;
            if ((unsigned)ih < (unsigned)H_ && (unsigned)iw < (unsigned)W_) {
                const float* src = (e2.y & (1 << 24)) ? Xi : Xr;
                v1 = src[basePix + (e2.y & 0xffff) + n];
            }
        }
        bf16x2 p; p[0] = (__bf16)v0; p[1] = (__bf16)v1;
        *(bf16x2*)(Bw + n*ASTR_ + kk) = p;
    }
    __syncthreads();

    // ---- preload the four B fragments (shared by all channel tiles) ----
    const int mRow = lane & 15;        // A row / B column / D column
    const int hi   = lane >> 4;        // lane half
    const int aSel = hi << 3;          // A K-offset within 16-block: 0 or 8
    const int bSel = hi << 3;          // B K-offset within 16-block: 0 or 8

    bf16x16 bq[4];
    #pragma unroll
    for (int kb = 0; kb < 4; ++kb) {
        const __bf16* bp = Bw + mRow*ASTR_ + kb*32 + bSel;
        bf16x8 lo = *(const bf16x8*)(bp);
        bf16x8 hi8 = *(const bf16x8*)(bp + 16);
        #pragma unroll
        for (int i = 0; i < 8; ++i) { bq[kb][i] = lo[i]; bq[kb][i+8] = hi8[i]; }
    }

    const int outBase = (b*COUT_)*HW_ + h*W_ + w0 + mRow;  // + ch*HW_
    float2* Tw = TR + wave * 16 * TSTR_;

    #pragma unroll
    for (int ct = 0; ct < 4; ++ct) {
        v8f accR = {}; v8f accI = {};
        #pragma unroll
        for (int kb = 0; kb < 4; ++kb) {
            const __bf16* arp = Ar + (ct*16 + mRow)*ASTR_ + kb*32 + aSel;
            const __bf16* aip = Ai + (ct*16 + mRow)*ASTR_ + kb*32 + aSel;
            bf16x8 arLo = *(const bf16x8*)(arp);
            bf16x8 arHi = *(const bf16x8*)(arp + 16);
            bf16x8 aiLo = *(const bf16x8*)(aip);
            bf16x8 aiHi = *(const bf16x8*)(aip + 16);
            bf16x16 afr, afi;
            #pragma unroll
            for (int i = 0; i < 8; ++i) {
                afr[i] = arLo[i]; afr[i+8] = arHi[i];
                afi[i] = aiLo[i]; afi[i+8] = aiHi[i];
            }
            accR = __builtin_amdgcn_wmma_f32_16x16x32_bf16(
                       false, afr, false, bq[kb], (short)0, accR, false, false);
            accI = __builtin_amdgcn_wmma_f32_16x16x32_bf16(
                       false, afi, false, bq[kb], (short)0, accI, false, false);
        }
        // bias + global store + stage (yr,yi) into per-wave transpose buffer
        #pragma unroll
        for (int v = 0; v < 8; ++v) {
            int chl = v + (hi << 3);             // local channel 0..15
            int ch  = ct*16 + chl;
            float yr = accR[v] + br[ch];
            float yi = accI[v] + bi[ch];
            out[outBase + ch*HW_]             = yr;
            out[IMAG_OFF_ + outBase + ch*HW_] = yi;
            float2 pv; pv.x = yr; pv.y = yi;
            Tw[chl*TSTR_ + mRow] = pv;           // same-wave LDS, in-order
        }
        // wave-internal reduce: lane owns channel mRow, half hi (8 pixels)
        {
            float s0 = 0.f, q0 = 0.f, s1 = 0.f, q1 = 0.f;
            #pragma unroll
            for (int l = 0; l < 8; ++l) {
                float2 pv = Tw[mRow*TSTR_ + (hi << 3) + l];
                s0 += pv.x; q0 += pv.x*pv.x;
                s1 += pv.y; q1 += pv.y*pv.y;
            }
            s0 += __shfl_xor(s0, 16, 32);
            q0 += __shfl_xor(q0, 16, 32);
            s1 += __shfl_xor(s1, 16, 32);
            q1 += __shfl_xor(q1, 16, 32);
            if (hi == 0) {
                float* p = WS + (wave*COUT_ + ct*16 + mRow)*4;
                p[0] = s0; p[1] = q0; p[2] = s1; p[3] = q1;
            }
        }
    }
    __syncthreads();

    // ---- deterministic block reduction of wave partials ----
    {
        int ch = tid >> 2, st = tid & 3;
        float s = 0.f;
        #pragma unroll
        for (int w = 0; w < WAVES_; ++w) s += WS[(w*COUT_ + ch)*4 + st];
        partials[(size_t)blockIdx.x * 256 + tid] = s;   // [block][ch*4 + stat]
    }
}

// ============================================================================
// Pass 2: fold 4096 block partials -> per-channel BN scale/shift (deterministic)
// ============================================================================
extern "C" __global__ void bn_stats_finalize(
    const float* __restrict__ partials,
    const float* __restrict__ gamma_r, const float* __restrict__ beta_r,
    const float* __restrict__ gamma_i, const float* __restrict__ beta_i,
    float* __restrict__ coef)
{
    int t = threadIdx.x;               // 128 threads: {r,i} x 64 channels
    if (t >= 128) return;
    int part = t >> 6, ch = t & 63;
    float s = 0.f, q = 0.f;
    #pragma unroll 8
    for (int bk = 0; bk < BLOCKS1_; ++bk) {
        const float* p = partials + (size_t)bk*256 + ch*4 + part*2;
        s += p[0]; q += p[1];
    }
    const float invN = 1.0f / (float)NPIX_;
    float mean = s * invN;
    float var  = fmaxf(q * invN - mean*mean, 0.f);
    float rstd = rsqrtf(var + 1e-5f);
    float g  = part ? gamma_i[ch] : gamma_r[ch];
    float be = part ? beta_i[ch]  : beta_r[ch];
    float scale = g * rstd;
    coef[part*128 + ch]      = scale;
    coef[part*128 + 64 + ch] = be - mean*scale;
}

// ============================================================================
// Pass 3: streaming in-place normalize of d_out (float4, same ch per vec4)
// ============================================================================
extern "C" __global__ __launch_bounds__(256)
void bn_apply(float* __restrict__ out, const float* __restrict__ coef)
{
    int e4 = blockIdx.x * 256 + threadIdx.x;   // < 2^24
    int e  = e4 << 2;                          // element idx < 2^27
    int ch   = (e >> 14) & 63;                 // HW_ = 2^14
    int part = e >> 25;                        // IMAG_OFF_ = 2^25
    float scale = coef[part*128 + ch];
    float shift = coef[part*128 + 64 + ch];
    float4* p = (float4*)out + e4;
    float4 v = *p;
    v.x = v.x*scale + shift;
    v.y = v.y*scale + shift;
    v.z = v.z*scale + shift;
    v.w = v.w*scale + shift;
    *p = v;
}

// ============================================================================
extern "C" void kernel_launch(void* const* d_in, const int* in_sizes, int n_in,
                              void* d_out, int out_size, void* d_ws, size_t ws_size,
                              hipStream_t stream)
{
    const float* Xr = (const float*)d_in[0];
    const float* Xi = (const float*)d_in[1];
    const float* Wr = (const float*)d_in[2];
    const float* Wi = (const float*)d_in[3];
    const float* br = (const float*)d_in[4];
    const float* bi = (const float*)d_in[5];
    const float* gr = (const float*)d_in[6];
    const float* ber = (const float*)d_in[7];
    const float* gi = (const float*)d_in[8];
    const float* bei = (const float*)d_in[9];

    float* out      = (float*)d_out;
    float* partials = (float*)d_ws;                 // 4096*256 floats
    float* coef     = partials + COEF_OFF_;         // 256 floats

    size_t shmem = (size_t)(2*COUT_*ASTR_ + WAVES_*16*ASTR_) * sizeof(__bf16)
                 + (size_t)WAVES_*COUT_*4 * sizeof(float)
                 + (size_t)KRED_ * sizeof(int)
                 + (size_t)WAVES_*16*TSTR_ * sizeof(float2);   // 95,744 B

    cplx_conv_wmma<<<BLOCKS1_, 256, shmem, stream>>>(Xr, Xi, Wr, Wi, br, bi,
                                                     out, partials);
    bn_stats_finalize<<<1, 128, 0, stream>>>(partials, gr, ber, gi, bei, coef);

    const int nvec4 = (2*B_*COUT_*HW_) / 4;         // 16,777,216
    bn_apply<<<nvec4/256, 256, 0, stream>>>(out, coef);
}
